// GCN_1099511628226
// MI455X (gfx1250) — matrix operand order repaired and checked by reading, not compiled, not run on hardware
//
#include <hip/hip_runtime.h>
#include <hip/hip_bf16.h>

typedef __attribute__((ext_vector_type(16))) _Float16 v16h;
typedef __attribute__((ext_vector_type(8)))  _Float16 v8h;
typedef __attribute__((ext_vector_type(8)))  float    v8f;

#define GCN_N   100000
#define GCN_E   1000000
#define GCN_H   128
#define GCN_NC  40
#define GCN_NCP 48
#define GCN_EPS 1e-5f
#define GCN_MT  5              // M tiles per block -> 80 rows; 100000/80 = 1250 exact

// ---------------------------------------------------------------- utilities

__global__ void fill_zero_kernel(float* __restrict__ p, int n) {
    int i = blockIdx.x * blockDim.x + threadIdx.x;
    if (i < n) p[i] = 0.0f;
}

// degree counting: dout[src[e]] += 1 ; din[dst[e]] += 1
__global__ void degree_kernel(const int* __restrict__ src, const int* __restrict__ dst,
                              float* __restrict__ dout, float* __restrict__ din, int E) {
    int e = blockIdx.x * blockDim.x + threadIdx.x;
    if (e < E) {
        atomicAdd(&dout[src[e]], 1.0f);
        atomicAdd(&din[dst[e]], 1.0f);
    }
}

// Pack weight W[128 x cols] (row-major f32) into per-lane WMMA B fragments (f16),
// zero-padded to padcols. Fragment layout (16 contiguous halfs per lane):
//   Bp[(((ct*4 + kb)*32 + lane)*16) + j] = W[kb*32 + koB + j][ct*16 + lane%16]
//   koB = (lane < 16) ? 0 : 16
__global__ void pack_b_kernel(const float* __restrict__ W, _Float16* __restrict__ Bp,
                              int cols, int padcols) {
    int i = blockIdx.x * blockDim.x + threadIdx.x;
    int total = padcols * GCN_H;           // (padcols/16) * 4 * 32 * 16
    if (i >= total) return;
    int j    = i & 15;
    int lane = (i >> 4) & 31;
    int kb   = (i >> 9) & 3;
    int ct   = i >> 11;
    int koB  = (lane < 16) ? 0 : 16;
    int k    = kb * 32 + koB + j;
    int n    = ct * 16 + (lane & 15);
    Bp[i] = (n < cols) ? (_Float16)W[k * cols + n] : (_Float16)0.0f;
}

// ---------------------------------------------------------------- WMMA GEMM
// C[N x ncolsPad] = rowscale(A[N x 128]) @ W[128 x ncolsPad]   (W pre-packed as Bp)
// grid.x = N/80; blockDim.x = 32 * (ncolsPad/16). One wave per 16-col tile, 5 M-tiles.
__global__ void gemm_wmma_kernel(const float* __restrict__ A, const float* __restrict__ deg,
                                 const _Float16* __restrict__ Bp, float* __restrict__ C,
                                 int ncolsPad) {
    __shared__ __align__(16) _Float16 As[GCN_MT * 16 * GCN_H];   // 80 x 128 f16 = 20 KB

    const int tid = threadIdx.x;
    const int rowBlk = blockIdx.x * (GCN_MT * 16);

    // cooperative stage of 80x128 A tile into LDS as f16 (with out-degree row scaling)
    for (int i = tid; i < GCN_MT * 16 * GCN_H; i += blockDim.x) {
        int row = rowBlk + (i >> 7);          // always < N (N % 80 == 0)
        float v = A[(long long)row * GCN_H + (i & 127)];
        if (deg) v *= rsqrtf(fmaxf(deg[row], 1.0f));
        As[i] = (_Float16)v;
    }
    __syncthreads();

    const int lane = tid & 31;
    const int ct   = tid >> 5;                // col tile == wave id
    const int m    = lane & 15;
    const int nC   = ct * 16 + m;
    const int koA  = (lane < 16) ? 0 : 16;

    // B fragments: 4 K-steps, 16 contiguous halfs per lane -> 2x b128 each, reused 5x
    v16h b[4];
    const _Float16* bp = Bp + ((size_t)(ct * 4) * 32 + lane) * 16;
    #pragma unroll
    for (int kb = 0; kb < 4; ++kb) {
        v8h blo = *(const v8h*)(bp + (size_t)kb * 32 * 16);
        v8h bhi = *(const v8h*)(bp + (size_t)kb * 32 * 16 + 8);
        b[kb] = __builtin_shufflevector(blo, bhi, 0, 1, 2, 3, 4, 5, 6, 7,
                                                  8, 9, 10, 11, 12, 13, 14, 15);
    }

    #pragma unroll
    for (int mt = 0; mt < GCN_MT; ++mt) {
        const _Float16* as = &As[(mt * 16 + m) * GCN_H];
        v8f acc = {};
        #pragma unroll
        for (int kb = 0; kb < 4; ++kb) {
            const int k0 = kb * 32;
            v8h alo = *(const v8h*)(as + k0 + koA);        // K = k0+koA   .. +7
            v8h ahi = *(const v8h*)(as + k0 + koA + 8);    // K = k0+koA+8 .. +15
            v16h a = __builtin_shufflevector(alo, ahi, 0, 1, 2, 3, 4, 5, 6, 7,
                                                       8, 9, 10, 11, 12, 13, 14, 15);
            acc = __builtin_amdgcn_wmma_f32_16x16x32_f16(
                      /*neg_a=*/false, a, /*neg_b=*/false, b[kb],
                      /*c_mod=*/(short)0, acc, /*reuse_a=*/false, /*reuse_b=*/false);
        }
        // C/D layout: VGPR r, lane L -> M = r + 8*(L/16), N = L%16
        const int rowBase = rowBlk + mt * 16 + ((lane >> 4) << 3);
        #pragma unroll
        for (int r = 0; r < 8; ++r)
            C[(long long)(rowBase + r) * ncolsPad + nC] = acc[r];
    }
}

// ---------------------------------------------------------------- SpMM scatter
// acc[dst[e]] += coef * X[src[e]]  where coef = attn[attnIdx] * rsqrt(max(din[dst],1))
__global__ void scatter_kernel(const int* __restrict__ src, const int* __restrict__ dst,
                               const float* __restrict__ X, const float* __restrict__ din,
                               const float* __restrict__ attn, int attnIdx,
                               float* __restrict__ acc, int E, int ncols) {
    const int per = ncols >> 2;                       // float4 groups per edge
    int tid = blockIdx.x * blockDim.x + threadIdx.x;
    if (tid >= E * per) return;
    int e = tid / per;
    int g = tid - e * per;
    int s = src[e], d = dst[e];
    float coef = rsqrtf(fmaxf(din[d], 1.0f));
    if (attn) coef *= attn[attnIdx];
    const float4 x = *(const float4*)(X + (long long)s * ncols + g * 4);
    float* p = acc + (long long)d * ncols + g * 4;
    atomicAdd(p + 0, coef * x.x);
    atomicAdd(p + 1, coef * x.y);
    atomicAdd(p + 2, coef * x.z);
    atomicAdd(p + 3, coef * x.w);
}

// ---------------------------------------------------------------- combine + BN stats
// out = acc + lin ; per-column sum/sumsq (128 threads = 1 col each, 128 rows/block,
// one atomic per column per block)
__global__ void combine_stats_kernel(const float* __restrict__ acc, const float* __restrict__ lin,
                                     float* __restrict__ out,
                                     float* __restrict__ sum, float* __restrict__ sumsq,
                                     int nrows) {
    const int c = threadIdx.x;              // 0..127
    const int row0 = blockIdx.x * 128;
    float s = 0.0f, s2 = 0.0f;
    for (int r = 0; r < 128; ++r) {
        int row = row0 + r;
        if (row >= nrows) break;
        long long idx = (long long)row * GCN_H + c;
        float v = acc[idx] + lin[idx];
        out[idx] = v;
        s += v; s2 += v * v;
    }
    atomicAdd(&sum[c], s);
    atomicAdd(&sumsq[c], s2);
}

// BN (biased variance, training-mode math) + ReLU, in place
__global__ void bn_relu_kernel(float* __restrict__ h, const float* __restrict__ sum,
                               const float* __restrict__ sumsq,
                               const float* __restrict__ gamma, const float* __restrict__ beta,
                               int nrows, float invN) {
    int i = blockIdx.x * blockDim.x + threadIdx.x;
    if (i >= nrows * GCN_H) return;
    int c = i & (GCN_H - 1);
    float m = sum[c] * invN;
    float var = sumsq[c] * invN - m * m;
    float v = (h[i] - m) * rsqrtf(var + GCN_EPS) * gamma[c] + beta[c];
    h[i] = v > 0.0f ? v : 0.0f;
}

// final layer: out[N x 40] = acc48 + lin48 + b2 (strip 48 -> 40 padding)
__global__ void final_combine_kernel(const float* __restrict__ acc48, const float* __restrict__ lin48,
                                     const float* __restrict__ b2, float* __restrict__ out,
                                     int nrows) {
    int i = blockIdx.x * blockDim.x + threadIdx.x;
    if (i >= nrows * GCN_NC) return;
    int r = i / GCN_NC, c = i - r * GCN_NC;
    long long j = (long long)r * GCN_NCP + c;
    out[i] = acc48[j] + lin48[j] + b2[c];
}

// ---------------------------------------------------------------- launch

extern "C" void kernel_launch(void* const* d_in, const int* in_sizes, int n_in,
                              void* d_out, int out_size, void* d_ws, size_t ws_size,
                              hipStream_t stream) {
    const float* feat  = (const float*)d_in[0];
    const int*   src0  = (const int*)d_in[1];
    const int*   dst0  = (const int*)d_in[2];
    const int*   src1  = (const int*)d_in[3];
    const int*   dst1  = (const int*)d_in[4];
    const float* attn  = (const float*)d_in[5];
    const float* W0    = (const float*)d_in[6];
    const float* W1    = (const float*)d_in[7];
    const float* W2    = (const float*)d_in[8];
    const float* b2    = (const float*)d_in[9];
    const float* L0    = (const float*)d_in[10];
    const float* L1    = (const float*)d_in[11];
    const float* L2    = (const float*)d_in[12];
    const float* g0    = (const float*)d_in[13];
    const float* be0   = (const float*)d_in[14];
    const float* g1    = (const float*)d_in[15];
    const float* be1   = (const float*)d_in[16];
    float* out = (float*)d_out;

    const int N = GCN_N, E = GCN_E, H = GCN_H;

    // ---- workspace layout
    char* ws = (char*)d_ws;
    size_t off = 0;
    auto allocF = [&](size_t n) { float* p = (float*)(ws + off); off += n * sizeof(float); return p; };
    float* B1 = allocF((size_t)N * H);     // 51.2 MB each
    float* B2 = allocF((size_t)N * H);
    float* B3 = allocF((size_t)N * H);
    float* B4 = allocF((size_t)N * H);     // scatter accumulator
    float* dout0 = allocF(N);
    float* din0  = allocF(N);
    float* dout1 = allocF(N);
    float* din1  = allocF(N);
    float* sum   = allocF(H);
    float* sumsq = allocF(H);
    auto allocH = [&](size_t n) { _Float16* p = (_Float16*)(ws + off); off += n * sizeof(_Float16); return p; };
    _Float16* W0p = allocH((size_t)H * H);
    _Float16* W1p = allocH((size_t)H * H);
    _Float16* L0p = allocH((size_t)H * H);
    _Float16* L1p = allocH((size_t)H * H);
    _Float16* W2p = allocH((size_t)H * GCN_NCP);
    _Float16* L2p = allocH((size_t)H * GCN_NCP);

    auto cdiv = [](int a, int b) { return (a + b - 1) / b; };
    const float invN = 1.0f / (float)N;

    // ---- zero degrees + stats (contiguous: 4N + 256 floats) and accumulator
    fill_zero_kernel<<<cdiv(4 * N + 2 * H, 256), 256, 0, stream>>>(dout0, 4 * N + 2 * H);
    fill_zero_kernel<<<cdiv(N * H, 256), 256, 0, stream>>>(B4, N * H);

    // ---- degrees for both graphs
    degree_kernel<<<cdiv(E, 256), 256, 0, stream>>>(src0, dst0, dout0, din0, E);
    degree_kernel<<<cdiv(E, 256), 256, 0, stream>>>(src1, dst1, dout1, din1, E);

    // ---- weights: f32 -> f16 packed WMMA B fragments (zero padded for the 40-col layer)
    pack_b_kernel<<<cdiv(H * H, 256), 256, 0, stream>>>(W0, W0p, H, H);
    pack_b_kernel<<<cdiv(H * H, 256), 256, 0, stream>>>(W1, W1p, H, H);
    pack_b_kernel<<<cdiv(H * H, 256), 256, 0, stream>>>(L0, L0p, H, H);
    pack_b_kernel<<<cdiv(H * H, 256), 256, 0, stream>>>(L1, L1p, H, H);
    pack_b_kernel<<<cdiv(H * GCN_NCP, 256), 256, 0, stream>>>(W2, W2p, GCN_NC, GCN_NCP);
    pack_b_kernel<<<cdiv(H * GCN_NCP, 256), 256, 0, stream>>>(L2, L2p, GCN_NC, GCN_NCP);

    const int rowBlks = N / (GCN_MT * 16);   // 1250, exact
    // =============================== layer 0
    gemm_wmma_kernel<<<rowBlks, 256, 0, stream>>>(feat, dout0, W0p, B1, H);      // X0
    gemm_wmma_kernel<<<rowBlks, 256, 0, stream>>>(feat, dout1, W0p, B2, H);      // X1
    gemm_wmma_kernel<<<rowBlks, 256, 0, stream>>>(feat, nullptr, L0p, B3, H);    // feat @ L0
    scatter_kernel<<<cdiv(E * (H / 4), 256), 256, 0, stream>>>(src0, dst0, B1, din0, attn, 0, B4, E, H);
    scatter_kernel<<<cdiv(E * (H / 4), 256), 256, 0, stream>>>(src1, dst1, B2, din1, attn, 1, B4, E, H);
    combine_stats_kernel<<<cdiv(N, 128), 128, 0, stream>>>(B4, B3, B1, sum, sumsq, N);
    bn_relu_kernel<<<cdiv(N * H, 256), 256, 0, stream>>>(B1, sum, sumsq, g0, be0, N, invN);

    // =============================== layer 1 (graph0)
    fill_zero_kernel<<<cdiv(N * H, 256), 256, 0, stream>>>(B4, N * H);
    fill_zero_kernel<<<cdiv(2 * H, 256), 256, 0, stream>>>(sum, 2 * H);
    gemm_wmma_kernel<<<rowBlks, 256, 0, stream>>>(B1, dout0, W1p, B2, H);        // X
    gemm_wmma_kernel<<<rowBlks, 256, 0, stream>>>(B1, nullptr, L1p, B3, H);      // h1 @ L1
    scatter_kernel<<<cdiv(E * (H / 4), 256), 256, 0, stream>>>(src0, dst0, B2, din0, nullptr, 0, B4, E, H);
    combine_stats_kernel<<<cdiv(N, 128), 128, 0, stream>>>(B4, B3, B2, sum, sumsq, N);
    bn_relu_kernel<<<cdiv(N * H, 256), 256, 0, stream>>>(B2, sum, sumsq, g1, be1, N, invN);

    // =============================== layer 2 (graph0, 40 cols padded to 48)
    fill_zero_kernel<<<cdiv(N * GCN_NCP, 256), 256, 0, stream>>>(B4, N * GCN_NCP);
    gemm_wmma_kernel<<<rowBlks, 32 * (GCN_NCP / 16), 0, stream>>>(B2, dout0, W2p, B1, GCN_NCP);   // X2
    gemm_wmma_kernel<<<rowBlks, 32 * (GCN_NCP / 16), 0, stream>>>(B2, nullptr, L2p, B3, GCN_NCP); // h2 @ L2
    scatter_kernel<<<cdiv(E * (GCN_NCP / 4), 256), 256, 0, stream>>>(src0, dst0, B1, din0, nullptr, 0, B4, E, GCN_NCP);
    final_combine_kernel<<<cdiv(N * GCN_NC, 256), 256, 0, stream>>>(B4, B3, b2, out, N);
}
